// S2S_4063039062752
// MI455X (gfx1250) — matrix operand order repaired and checked by reading, not compiled, not run on hardware
//
#include <hip/hip_runtime.h>
#include <math.h>

#define Nn 8192
#define Kn 32
#define Dd 256
#define Pp 64
#define Hh 8
#define DHh 32
#define NBATCH 32
#define NCHAIN 64
#define RBF_BINS 16

typedef __attribute__((ext_vector_type(16))) __bf16 v16bf;
typedef __attribute__((ext_vector_type(8)))  __bf16 v8bf;
typedef __attribute__((ext_vector_type(8)))  float  v8f;

// ---------------------------------------------------------------------------
// helpers
// ---------------------------------------------------------------------------
// branchless tanh-gelu: tanh(y) = sign(y)*(1-e)/(1+e), e = exp(-2|y|)
__device__ inline float gelu_f(float x) {
    const float c = 0.7978845608028654f; // sqrt(2/pi)
    float y = c * (x + 0.044715f * x * x * x);
    float e = __expf(-2.f * fabsf(y));
    float t = (1.f - e) / (1.f + e);
    t = copysignf(t, y);
    return 0.5f * x * (1.f + t);
}

__device__ inline float warp_max(float v) {
#pragma unroll
    for (int s = 16; s > 0; s >>= 1) v = fmaxf(v, __shfl_xor(v, s, 32));
    return v;
}
__device__ inline float warp_sum(float v) {
#pragma unroll
    for (int s = 16; s > 0; s >>= 1) v += __shfl_xor(v, s, 32);
    return v;
}

// A operand (16x32 bf16, M x K), row-major A with leading dim lda.
// ISA layout: lanes 0-15: row=lane, K = {0..7, 16..23}; lanes 16-31: row=lane-16, K = {8..15, 24..31}
__device__ inline v16bf load_a_bf16(const __bf16* A, int lda, int m0, int k0) {
    int lane = threadIdx.x & 31;
    int row  = m0 + (lane & 15);
    int kk   = k0 + ((lane < 16) ? 0 : 8);
    const __bf16* p = A + (size_t)row * lda + kk;
    v8bf lo = *(const v8bf*)p;
    v8bf hi = *(const v8bf*)(p + 16);
    v16bf r;
#pragma unroll
    for (int i = 0; i < 8; i++) { r[i] = lo[i]; r[i + 8] = hi[i]; }
    return r;
}

// B operand (32x16 bf16, K x N) sourced from transposed weights Bt[n][k], leading dim ldb.
// ISA layout: lanes 0-15: col=lane, K = {0..7, 16..23}; lanes 16-31: col=lane-16, K = {8..15, 24..31}
__device__ inline v16bf load_b_bf16(const __bf16* Bt, int ldb, int n0, int k0) {
    int lane = threadIdx.x & 31;
    int col  = n0 + (lane & 15);
    int kk   = k0 + ((lane < 16) ? 0 : 8);
    const __bf16* p = Bt + (size_t)col * ldb + kk;
    v8bf lo = *(const v8bf*)p;
    v8bf hi = *(const v8bf*)(p + 16);
    v16bf r;
#pragma unroll
    for (int i = 0; i < 8; i++) { r[i] = lo[i]; r[i + 8] = hi[i]; }
    return r;
}

__device__ inline v8f wmma_bf(v16bf a, v16bf b, v8f c) {
    return __builtin_amdgcn_wmma_f32_16x16x32_bf16(false, a, false, b, (short)0, c, false, false);
}

// frames: e1=norm(c-ca); e2=norm(u - (u.e1)e1), u=n-ca; e3=e1 x e2; R[:,0]=e1,[:,1]=e2,[:,2]=e3; t=ca
__device__ inline void compute_frame(const float* __restrict__ pos, int idx,
                                     float R[3][3], float t[3]) {
    const float* p = pos + (size_t)idx * 15;
    float ca[3] = {p[3], p[4], p[5]};
    float e1[3], e2[3], u[3];
    float s = 0.f;
#pragma unroll
    for (int i = 0; i < 3; i++) { e1[i] = p[6 + i] - ca[i]; s += e1[i] * e1[i]; }
    float inv = rsqrtf(s + 1e-8f);
#pragma unroll
    for (int i = 0; i < 3; i++) e1[i] *= inv;
    float d1 = 0.f;
#pragma unroll
    for (int i = 0; i < 3; i++) { u[i] = p[i] - ca[i]; d1 += u[i] * e1[i]; }
    s = 0.f;
#pragma unroll
    for (int i = 0; i < 3; i++) { e2[i] = u[i] - d1 * e1[i]; s += e2[i] * e2[i]; }
    inv = rsqrtf(s + 1e-8f);
#pragma unroll
    for (int i = 0; i < 3; i++) e2[i] *= inv;
    float e3[3] = {e1[1] * e2[2] - e1[2] * e2[1],
                   e1[2] * e2[0] - e1[0] * e2[2],
                   e1[0] * e2[1] - e1[1] * e2[0]};
#pragma unroll
    for (int a = 0; a < 3; a++) { R[a][0] = e1[a]; R[a][1] = e2[a]; R[a][2] = e3[a]; t[a] = ca[a]; }
}

// ---------------------------------------------------------------------------
// prep kernels
// ---------------------------------------------------------------------------
__global__ void transpose_bf16_kernel(const float* __restrict__ W, __bf16* __restrict__ Bt,
                                      int Kc, int Nc) {
    int idx = blockIdx.x * 256 + threadIdx.x;
    if (idx >= Kc * Nc) return;
    int k = idx / Nc, n = idx - k * Nc;
    Bt[(size_t)n * Kc + k] = (__bf16)W[idx];
}

__global__ void convert_bf16_kernel(const float* __restrict__ X, __bf16* __restrict__ Y, int total) {
    int idx = blockIdx.x * 256 + threadIdx.x;
    if (idx < total) Y[idx] = (__bf16)X[idx];
}

__global__ void zero_kernel(float* __restrict__ p, int total) {
    int idx = blockIdx.x * 256 + threadIdx.x;
    if (idx < total) p[idx] = 0.f;
}

// ---------------------------------------------------------------------------
// generic WMMA GEMM: C = act(A[MxK] @ B[KxN] + bias), A bf16 row-major, Bt bf16 (NxK).
// Each wave owns a 16x64 output strip: 4 accumulators, one A-operand fetch per k-step.
// The A 16x32 tile is streamed into per-wave LDS with CDNA5 async-to-LDS copies,
// double-buffered (s_wait_asynccnt pipelining); B comes straight from L2-resident weights.
// Wave-to-tile mapping is flattened so all 8 waves stay busy for any Ncols in {256,512}.
// ---------------------------------------------------------------------------
template <int ACT>
__global__ __launch_bounds__(256) void gemm_kernel(
    const __bf16* __restrict__ A, const __bf16* __restrict__ Bt,
    const float* __restrict__ bias, float* __restrict__ Cf, __bf16* __restrict__ Cb,
    int M, int Ncols, int Kc) {
    __shared__ __attribute__((aligned(16))) __bf16 sA[8][2][512]; // 16 rows x 32 k, double buffered
    int wave = threadIdx.x >> 5;
    int lane = threadIdx.x & 31;
    int ntn = Ncols >> 6;                 // 64-col tiles per row band
    int wid = blockIdx.x * 8 + wave;
    int mt = wid / ntn;
    int nt = wid - mt * ntn;
    int m0 = mt * 16;
    int n0 = nt * 64;
    if (m0 >= M) return;

    unsigned sbase0 = (unsigned)(size_t)(&sA[wave][0][0]);
    unsigned sbase1 = (unsigned)(size_t)(&sA[wave][1][0]);

    // async stage: 16 rows x 32 bf16 = 64 chunks of 16B, 32 lanes x 2 issues
    auto stage = [&](int k0, unsigned sbase) {
#pragma unroll
        for (int c = 0; c < 2; c++) {
            int q = c * 32 + lane;
            int r = q >> 2, part = q & 3;
            const __bf16* g = A + (size_t)(m0 + r) * Kc + k0 + part * 8;
            unsigned ldsoff = sbase + q * 16;
            asm volatile("global_load_async_to_lds_b128 %0, %1, off"
                         :: "v"(ldsoff), "v"(g) : "memory");
        }
    };

    v8f acc0 = {}, acc1 = {}, acc2 = {}, acc3 = {};
    int nk = Kc >> 5;
    stage(0, sbase0);
    for (int i = 0; i < nk; i++) {
        if (i + 1 < nk) {
            stage((i + 1) << 5, (i & 1) ? sbase0 : sbase1);
            __builtin_prefetch(Bt + (size_t)(n0 + (lane & 15)) * Kc + ((i + 1) << 5), 0, 1);
            asm volatile("s_wait_asynccnt 2" ::: "memory"); // buffer i complete, i+1 in flight
        } else {
            asm volatile("s_wait_asynccnt 0" ::: "memory");
        }
        const __bf16* ap = (i & 1) ? (const __bf16*)&sA[wave][1][0]
                                   : (const __bf16*)&sA[wave][0][0];
        v16bf a = load_a_bf16(ap, 32, 0, 0);
        int k0 = i << 5;
        acc0 = wmma_bf(a, load_b_bf16(Bt, Kc, n0,      k0), acc0);
        acc1 = wmma_bf(a, load_b_bf16(Bt, Kc, n0 + 16, k0), acc1);
        acc2 = wmma_bf(a, load_b_bf16(Bt, Kc, n0 + 32, k0), acc2);
        acc3 = wmma_bf(a, load_b_bf16(Bt, Kc, n0 + 48, k0), acc3);
    }

    int rbase = m0 + ((lane < 16) ? 0 : 8);
    auto epil = [&](v8f acc, int colbase) {
        int col = colbase + (lane & 15);
        float bv = bias ? bias[col] : 0.f;
#pragma unroll
        for (int i2 = 0; i2 < 8; i2++) {
            float v = acc[i2] + bv;
            if (ACT == 1) v = gelu_f(v);
            size_t o = (size_t)(rbase + i2) * Ncols + col;
            if (Cf) Cf[o] = v;
            if (Cb) Cb[o] = (__bf16)v;
        }
    };
    epil(acc0, n0);
    epil(acc1, n0 + 16);
    epil(acc2, n0 + 32);
    epil(acc3, n0 + 48);
}

// ---------------------------------------------------------------------------
// pair branch: one wave per node. Features -> LN -> MLP(64->128->64) via WMMA -> attn bias
// ---------------------------------------------------------------------------
__global__ __launch_bounds__(32) void pair_kernel(
    const float* __restrict__ pos, const int* __restrict__ nbr,
    const int* __restrict__ resi, const int* __restrict__ chain,
    const unsigned char* __restrict__ mask,
    const float* __restrict__ w_relpos, const float* __restrict__ w_dist,
    const float* __restrict__ w_dir, const float* __restrict__ w_rot,
    const float* __restrict__ ln_g, const float* __restrict__ ln_b,
    const __bf16* __restrict__ wp1T, const float* __restrict__ b_p1,
    const __bf16* __restrict__ wp2T, const float* __restrict__ b_p2,
    const float* __restrict__ wb,
    float* __restrict__ biasL, float* __restrict__ pmask) {
    __shared__ __attribute__((aligned(16))) __bf16 pa[32][64];
    __shared__ __attribute__((aligned(16))) __bf16 mid[32][128];
    __shared__ __attribute__((aligned(16))) float  p2[32][64];

    int n = blockIdx.x;
    int lane = threadIdx.x;
    int j0 = nbr[n * Kn + lane];
    float validf = (j0 >= 0) ? 1.f : 0.f;
    int j = (j0 < 0) ? 0 : j0;
    float pm = (mask[n] ? 1.f : 0.f) * (mask[j] ? 1.f : 0.f) * validf;
    pmask[n * Kn + lane] = pm;

    float sc = (chain[j] == chain[n]) ? 1.f : 0.f;
    int off = resi[j] - resi[n];
    off = off < -32 ? -32 : (off > 32 ? 32 : off);
    off += 32;

    // RBF of CB distance
    float dx = pos[j * 15 + 12] - pos[n * 15 + 12];
    float dy = pos[j * 15 + 13] - pos[n * 15 + 13];
    float dz = pos[j * 15 + 14] - pos[n * 15 + 14];
    float dcb = sqrtf(dx * dx + dy * dy + dz * dz + 1e-8f);
    float rb[RBF_BINS];
#pragma unroll
    for (int b = 0; b < RBF_BINS; b++) {
        float c = (22.f / 15.f) * b;
        float z = (dcb - c) * (16.f / 22.f);
        rb[b] = __expf(-z * z);
    }
    // directions to 5 atoms
    float can[3] = {pos[n * 15 + 3], pos[n * 15 + 4], pos[n * 15 + 5]};
    float dirs[15];
#pragma unroll
    for (int a = 0; a < 5; a++) {
        float v0 = pos[j * 15 + a * 3 + 0] - can[0];
        float v1 = pos[j * 15 + a * 3 + 1] - can[1];
        float v2 = pos[j * 15 + a * 3 + 2] - can[2];
        float inv = rsqrtf(v0 * v0 + v1 * v1 + v2 * v2 + 1e-8f);
        dirs[a * 3 + 0] = v0 * inv; dirs[a * 3 + 1] = v1 * inv; dirs[a * 3 + 2] = v2 * inv;
    }
    // relative frames
    float Rn[3][3], tn[3], Rj[3][3], tj[3];
    compute_frame(pos, n, Rn, tn);
    compute_frame(pos, j, Rj, tj);
    float feat[12];
#pragma unroll
    for (int b = 0; b < 3; b++)
#pragma unroll
        for (int c = 0; c < 3; c++)
            feat[b * 3 + c] = Rn[0][b] * Rj[0][c] + Rn[1][b] * Rj[1][c] + Rn[2][b] * Rj[2][c];
    float dt0 = tj[0] - tn[0], dt1 = tj[1] - tn[1], dt2 = tj[2] - tn[2];
#pragma unroll
    for (int b = 0; b < 3; b++)
        feat[9 + b] = Rn[0][b] * dt0 + Rn[1][b] * dt1 + Rn[2][b] * dt2;

    // accumulate 64 pair features, LayerNorm
    float row[64];
    float mean = 0.f;
    for (int p = 0; p < 64; p++) {
        float acc = w_relpos[off * 64 + p] * sc;
        for (int b = 0; b < RBF_BINS; b++) acc += rb[b] * w_dist[b * 64 + p];
        for (int i = 0; i < 15; i++) acc += dirs[i] * w_dir[i * 64 + p];
        for (int i = 0; i < 12; i++) acc += feat[i] * w_rot[i * 64 + p];
        row[p] = acc;
        mean += acc;
    }
    mean *= (1.f / 64.f);
    float var = 0.f;
    for (int p = 0; p < 64; p++) { float d = row[p] - mean; var += d * d; }
    var *= (1.f / 64.f);
    float inv = rsqrtf(var + 1e-5f);
    for (int p = 0; p < 64; p++)
        pa[lane][p] = (__bf16)((row[p] - mean) * inv * ln_g[p] + ln_b[p]);
    __syncthreads();

    const __bf16* pap = (const __bf16*)pa;
    // MLP layer 1: 64 -> 128, gelu
    for (int mt = 0; mt < 2; mt++) {
        for (int nt = 0; nt < 8; nt++) {
            v8f acc = {};
            acc = wmma_bf(load_a_bf16(pap, 64, mt * 16, 0),  load_b_bf16(wp1T, 64, nt * 16, 0),  acc);
            acc = wmma_bf(load_a_bf16(pap, 64, mt * 16, 32), load_b_bf16(wp1T, 64, nt * 16, 32), acc);
            int col = nt * 16 + (lane & 15);
            float bv = b_p1[col];
            int rbase = mt * 16 + ((lane < 16) ? 0 : 8);
#pragma unroll
            for (int i = 0; i < 8; i++) mid[rbase + i][col] = (__bf16)gelu_f(acc[i] + bv);
        }
    }
    __syncthreads();
    const __bf16* midp = (const __bf16*)mid;
    // MLP layer 2: 128 -> 64
    for (int mt = 0; mt < 2; mt++) {
        for (int nt = 0; nt < 4; nt++) {
            v8f acc = {};
#pragma unroll
            for (int k0 = 0; k0 < 128; k0 += 32)
                acc = wmma_bf(load_a_bf16(midp, 128, mt * 16, k0), load_b_bf16(wp2T, 128, nt * 16, k0), acc);
            int col = nt * 16 + (lane & 15);
            float bv = b_p2[col];
            int rbase = mt * 16 + ((lane < 16) ? 0 : 8);
#pragma unroll
            for (int i = 0; i < 8; i++) p2[rbase + i][col] = acc[i] + bv;
        }
    }
    __syncthreads();
    // attention bias: (pair @ wb), wb is 64x8
    for (int h = 0; h < Hh; h++) {
        float s = 0.f;
        for (int p = 0; p < 64; p++) s += p2[lane][p] * wb[p * Hh + h];
        biasL[(size_t)(n * Kn + lane) * Hh + h] = s;
    }
}

// ---------------------------------------------------------------------------
// attention: wave per node (8 waves per block)
// ---------------------------------------------------------------------------
__global__ __launch_bounds__(256) void attn_kernel(
    const float* __restrict__ qf, const float* __restrict__ kf, const float* __restrict__ vf,
    const float* __restrict__ biasL, const float* __restrict__ pmask,
    const int* __restrict__ nbr, __bf16* __restrict__ ob) {
    __shared__ float attn_s[8][32];
    __shared__ int   nb_s[8][32];
    int wave = threadIdx.x >> 5, lane = threadIdx.x & 31;
    int n = blockIdx.x * 8 + wave;
    int j = nbr[n * Kn + lane];
    if (j < 0) j = 0;
    nb_s[wave][lane] = j;
    float pm = pmask[n * Kn + lane];
    const float scale = 0.17677669529663687f; // 1/sqrt(32)
    for (int h = 0; h < Hh; h++) {
        float dot = 0.f;
        const float* qp = qf + (size_t)n * Dd + h * DHh;
        const float* kp = kf + (size_t)j * Dd + h * DHh;
#pragma unroll
        for (int d = 0; d < DHh; d++) dot += qp[d] * kp[d];
        float lg = dot * scale + biasL[(size_t)(n * Kn + lane) * Hh + h];
        lg = (pm > 0.f) ? lg : -1e9f;
        float mx = warp_max(lg);
        float e = __expf(lg - mx);
        float sm = warp_sum(e);
        attn_s[wave][lane] = e / sm;
        __syncthreads();
        float acc = 0.f; // lane == output dim
        for (int k = 0; k < Kn; k++)
            acc += attn_s[wave][k] * vf[(size_t)nb_s[wave][k] * Dd + h * DHh + lane];
        ob[(size_t)n * Dd + h * DHh + lane] = (__bf16)acc;
        __syncthreads();
    }
}

// ---------------------------------------------------------------------------
// x1 = LN(local + y*mask); inc = local + y*mask
// ---------------------------------------------------------------------------
__global__ __launch_bounds__(256) void addln_kernel(
    const float* __restrict__ xin, const float* __restrict__ yf,
    const unsigned char* __restrict__ mask,
    const float* __restrict__ g, const float* __restrict__ b,
    float* __restrict__ x1, __bf16* __restrict__ x1b, float* __restrict__ incf) {
    __shared__ float red[256];
    int n = blockIdx.x, t = threadIdx.x;
    float v = xin[(size_t)n * Dd + t] + yf[(size_t)n * Dd + t] * (mask[n] ? 1.f : 0.f);
    incf[(size_t)n * Dd + t] = v;
    red[t] = v; __syncthreads();
    for (int s = 128; s > 0; s >>= 1) { if (t < s) red[t] += red[t + s]; __syncthreads(); }
    float m = red[0] * (1.f / Dd); __syncthreads();
    float d = v - m;
    red[t] = d * d; __syncthreads();
    for (int s = 128; s > 0; s >>= 1) { if (t < s) red[t] += red[t + s]; __syncthreads(); }
    float var = red[0] * (1.f / Dd);
    float o = d * rsqrtf(var + 1e-5f) * g[t] + b[t];
    x1[(size_t)n * Dd + t] = o;
    x1b[(size_t)n * Dd + t] = (__bf16)o;
}

// ---------------------------------------------------------------------------
// tfeat raw features: wave per node, 19 per neighbour (16 rbf + 3 dir), * pair_mask
// ---------------------------------------------------------------------------
__global__ __launch_bounds__(32) void tfeat_kernel(
    const float* __restrict__ pos, const int* __restrict__ nbr,
    const float* __restrict__ pmask, __bf16* __restrict__ tfr) {
    int n = blockIdx.x, lane = threadIdx.x;
    float Rn[3][3], tn[3];
    compute_frame(pos, n, Rn, tn);
    int j = nbr[n * Kn + lane];
    if (j < 0) j = 0;
    float d0 = pos[j * 15 + 3] - tn[0];
    float d1 = pos[j * 15 + 4] - tn[1];
    float d2 = pos[j * 15 + 5] - tn[2];
    float rel[3];
#pragma unroll
    for (int b = 0; b < 3; b++) rel[b] = Rn[0][b] * d0 + Rn[1][b] * d1 + Rn[2][b] * d2;
    float s2 = rel[0] * rel[0] + rel[1] * rel[1] + rel[2] * rel[2];
    float dd = sqrtf(s2 + 1e-8f);
    float invn = rsqrtf(s2 + 1e-8f);
    float pm = pmask[n * Kn + lane];
    size_t base = (size_t)n * (Kn * (RBF_BINS + 3)) + lane * (RBF_BINS + 3);
#pragma unroll
    for (int b = 0; b < RBF_BINS; b++) {
        float c = (22.f / 15.f) * b;
        float z = (dd - c) * (16.f / 22.f);
        tfr[base + b] = (__bf16)(__expf(-z * z) * pm);
    }
#pragma unroll
    for (int c = 0; c < 3; c++) tfr[base + RBF_BINS + c] = (__bf16)(rel[c] * invn * pm);
}

// ---------------------------------------------------------------------------
// segment means (batch, chain) via global f32 atomics
// ---------------------------------------------------------------------------
__global__ __launch_bounds__(256) void seg_accum_kernel(
    const float* __restrict__ upf, const int* __restrict__ batch, const int* __restrict__ chain,
    const unsigned char* __restrict__ mask,
    float* __restrict__ sumB, float* __restrict__ cntB,
    float* __restrict__ sumC, float* __restrict__ cntC) {
    int n = blockIdx.x, t = threadIdx.x;
    float mf = mask[n] ? 1.f : 0.f;
    int b = batch[n], c = chain[n];
    for (int d = t; d < 2 * Dd; d += 256) {
        float v = upf[(size_t)n * (2 * Dd) + d] * mf;
        atomicAdd(&sumB[(size_t)b * (2 * Dd) + d], v);
        atomicAdd(&sumC[(size_t)c * (2 * Dd) + d], v);
    }
    if (t == 0) { atomicAdd(&cntB[b], mf); atomicAdd(&cntC[c], mf); }
}

__global__ void seg_div_kernel(float* __restrict__ sumB, const float* __restrict__ cntB,
                               float* __restrict__ sumC, const float* __restrict__ cntC) {
    int idx = blockIdx.x * 256 + threadIdx.x;
    if (idx < NBATCH * 512) {
        sumB[idx] *= (1.f / fmaxf(cntB[idx >> 9], 1.f));
    } else {
        int i = idx - NBATCH * 512;
        if (i < NCHAIN * 512) sumC[i] *= (1.f / fmaxf(cntC[i >> 9], 1.f));
    }
}

__global__ void gate_combine_kernel(
    const float* __restrict__ upf, const float* __restrict__ lgf,
    const float* __restrict__ cgf, const float* __restrict__ bgf,
    const float* __restrict__ sumB, const float* __restrict__ sumC,
    const int* __restrict__ batch, const int* __restrict__ chain,
    __bf16* __restrict__ hb) {
    int idx = blockIdx.x * 256 + threadIdx.x; // N*512
    int n = idx >> 9, d = idx & 511;
    float h = bgf[idx] * sumB[(size_t)batch[n] * 512 + d] +
              cgf[idx] * sumC[(size_t)chain[n] * 512 + d] +
              lgf[idx] * upf[idx];
    hb[idx] = (__bf16)h;
}

// ---------------------------------------------------------------------------
// final: y2 = y2g + tfo*mask; x2 = LN(x1+y2); out = x2 + LN(inc+y2)
// ---------------------------------------------------------------------------
__global__ __launch_bounds__(256) void final_kernel(
    const float* __restrict__ x1, const float* __restrict__ y2g, const float* __restrict__ tfo,
    const unsigned char* __restrict__ mask, const float* __restrict__ incf,
    const float* __restrict__ gu, const float* __restrict__ bu,
    const float* __restrict__ gf2, const float* __restrict__ bf2,
    float* __restrict__ out) {
    __shared__ float red[256];
    int n = blockIdx.x, t = threadIdx.x;
    float mf = mask[n] ? 1.f : 0.f;
    float y2 = y2g[(size_t)n * Dd + t] + tfo[(size_t)n * Dd + t] * mf;
    float xv = x1[(size_t)n * Dd + t] + y2;
    float iv = incf[(size_t)n * Dd + t] + y2;
    // LN(xv)
    red[t] = xv; __syncthreads();
    for (int s = 128; s > 0; s >>= 1) { if (t < s) red[t] += red[t + s]; __syncthreads(); }
    float m1 = red[0] * (1.f / Dd); __syncthreads();
    float d1 = xv - m1;
    red[t] = d1 * d1; __syncthreads();
    for (int s = 128; s > 0; s >>= 1) { if (t < s) red[t] += red[t + s]; __syncthreads(); }
    float x2v = d1 * rsqrtf(red[0] * (1.f / Dd) + 1e-5f) * gu[t] + bu[t];
    __syncthreads();
    // LN(iv)
    red[t] = iv; __syncthreads();
    for (int s = 128; s > 0; s >>= 1) { if (t < s) red[t] += red[t + s]; __syncthreads(); }
    float m2 = red[0] * (1.f / Dd); __syncthreads();
    float d2 = iv - m2;
    red[t] = d2 * d2; __syncthreads();
    for (int s = 128; s > 0; s >>= 1) { if (t < s) red[t] += red[t + s]; __syncthreads(); }
    float lnv = d2 * rsqrtf(red[0] * (1.f / Dd) + 1e-5f) * gf2[t] + bf2[t];
    out[(size_t)n * Dd + t] = x2v + lnv;
}

// ---------------------------------------------------------------------------
// host
// ---------------------------------------------------------------------------
extern "C" void kernel_launch(void* const* d_in, const int* in_sizes, int n_in,
                              void* d_out, int out_size, void* d_ws, size_t ws_size,
                              hipStream_t stream) {
    (void)in_sizes; (void)n_in; (void)out_size; (void)ws_size;
    const float* local   = (const float*)d_in[0];
    const float* pos     = (const float*)d_in[1];
    const int*   nbr     = (const int*)d_in[2];
    const int*   resi    = (const int*)d_in[3];
    const int*   chain   = (const int*)d_in[4];
    const int*   batch   = (const int*)d_in[5];
    const unsigned char* mask = (const unsigned char*)d_in[6];
    const float* w_relpos = (const float*)d_in[7];
    const float* w_dist  = (const float*)d_in[8];
    const float* w_dir   = (const float*)d_in[9];
    const float* w_rot   = (const float*)d_in[10];
    const float* ln_pair_g = (const float*)d_in[11];
    const float* ln_pair_b = (const float*)d_in[12];
    const float* w_p1    = (const float*)d_in[13];
    const float* b_p1    = (const float*)d_in[14];
    const float* w_p2    = (const float*)d_in[15];
    const float* b_p2    = (const float*)d_in[16];
    const float* wq      = (const float*)d_in[17];
    const float* wk      = (const float*)d_in[18];
    const float* wv      = (const float*)d_in[19];
    const float* wb      = (const float*)d_in[20];
    const float* wo      = (const float*)d_in[21];
    const float* ln_a_g  = (const float*)d_in[22];
    const float* ln_a_b  = (const float*)d_in[23];
    const float* w_t1    = (const float*)d_in[24];
    const float* w_t2    = (const float*)d_in[25];
    const float* w_up    = (const float*)d_in[26];
    const float* w_lg    = (const float*)d_in[27];
    const float* w_cg    = (const float*)d_in[28];
    const float* w_bg    = (const float*)d_in[29];
    const float* w_out   = (const float*)d_in[30];
    const float* b_out   = (const float*)d_in[31];
    const float* ln_u_g  = (const float*)d_in[32];
    const float* ln_u_b  = (const float*)d_in[33];
    const float* ln_f_g  = (const float*)d_in[34];
    const float* ln_f_b  = (const float*)d_in[35];
    float* outp = (float*)d_out;

    char* base = (char*)d_ws;
    size_t off = 0;
    auto alloc = [&](size_t bytes) -> void* {
        off = (off + 255) & ~(size_t)255;
        void* p = base + off;
        off += bytes;
        return p;
    };

    __bf16* xb    = (__bf16*)alloc((size_t)Nn * Dd * 2);
    __bf16* wqT   = (__bf16*)alloc(256 * 256 * 2);
    __bf16* wkT   = (__bf16*)alloc(256 * 256 * 2);
    __bf16* wvT   = (__bf16*)alloc(256 * 256 * 2);
    __bf16* woT   = (__bf16*)alloc(256 * 256 * 2);
    __bf16* wp1T  = (__bf16*)alloc(128 * 64 * 2);
    __bf16* wp2T  = (__bf16*)alloc(64 * 128 * 2);
    __bf16* wt1T  = (__bf16*)alloc(512 * 608 * 2);
    __bf16* wt2T  = (__bf16*)alloc(256 * 512 * 2);
    __bf16* wupT  = (__bf16*)alloc(512 * 256 * 2);
    __bf16* wlgT  = (__bf16*)alloc(512 * 256 * 2);
    __bf16* wcgT  = (__bf16*)alloc(512 * 256 * 2);
    __bf16* wbgT  = (__bf16*)alloc(512 * 256 * 2);
    __bf16* woutT = (__bf16*)alloc(256 * 512 * 2);
    float* qf    = (float*)alloc((size_t)Nn * Dd * 4);
    float* kf    = (float*)alloc((size_t)Nn * Dd * 4);
    float* vf    = (float*)alloc((size_t)Nn * Dd * 4);
    float* biasL = (float*)alloc((size_t)Nn * Kn * Hh * 4);
    float* pmask = (float*)alloc((size_t)Nn * Kn * 4);
    __bf16* ob   = (__bf16*)alloc((size_t)Nn * Dd * 2);
    float* yf    = (float*)alloc((size_t)Nn * Dd * 4);
    float* x1    = (float*)alloc((size_t)Nn * Dd * 4);
    __bf16* x1b  = (__bf16*)alloc((size_t)Nn * Dd * 2);
    float* incf  = (float*)alloc((size_t)Nn * Dd * 4);
    __bf16* tfr  = (__bf16*)alloc((size_t)Nn * 608 * 2);
    __bf16* t1b  = (__bf16*)alloc((size_t)Nn * 512 * 2);
    float* tfo   = (float*)alloc((size_t)Nn * Dd * 4);
    float* upf   = (float*)alloc((size_t)Nn * 512 * 4);
    float* lgf   = (float*)alloc((size_t)Nn * 512 * 4);
    float* cgf   = (float*)alloc((size_t)Nn * 512 * 4);
    float* bgf   = (float*)alloc((size_t)Nn * 512 * 4);
    __bf16* hb   = (__bf16*)alloc((size_t)Nn * 512 * 2);
    float* y2g   = (float*)alloc((size_t)Nn * Dd * 4);
    int segTotal = NBATCH * 512 + NBATCH + NCHAIN * 512 + NCHAIN;
    float* segbuf = (float*)alloc((size_t)segTotal * 4);
    float* sumB = segbuf;
    float* cntB = sumB + NBATCH * 512;
    float* sumC = cntB + NBATCH;
    float* cntC = sumC + NCHAIN * 512;

    // GEMM grids: waves flattened over (M/16) x (Ncols/64) tiles, 8 waves per block
    const int g256 = (Nn / 16) * (256 / 64) / 8; // 256 blocks
    const int g512 = (Nn / 16) * (512 / 64) / 8; // 512 blocks

    // -------- weight prep (idempotent, runs every call) --------
    transpose_bf16_kernel<<<(256 * 256 + 255) / 256, 256, 0, stream>>>(wq, wqT, 256, 256);
    transpose_bf16_kernel<<<(256 * 256 + 255) / 256, 256, 0, stream>>>(wk, wkT, 256, 256);
    transpose_bf16_kernel<<<(256 * 256 + 255) / 256, 256, 0, stream>>>(wv, wvT, 256, 256);
    transpose_bf16_kernel<<<(256 * 256 + 255) / 256, 256, 0, stream>>>(wo, woT, 256, 256);
    transpose_bf16_kernel<<<(64 * 128 + 255) / 256, 256, 0, stream>>>(w_p1, wp1T, 64, 128);
    transpose_bf16_kernel<<<(128 * 64 + 255) / 256, 256, 0, stream>>>(w_p2, wp2T, 128, 64);
    transpose_bf16_kernel<<<(608 * 512 + 255) / 256, 256, 0, stream>>>(w_t1, wt1T, 608, 512);
    transpose_bf16_kernel<<<(512 * 256 + 255) / 256, 256, 0, stream>>>(w_t2, wt2T, 512, 256);
    transpose_bf16_kernel<<<(256 * 512 + 255) / 256, 256, 0, stream>>>(w_up, wupT, 256, 512);
    transpose_bf16_kernel<<<(256 * 512 + 255) / 256, 256, 0, stream>>>(w_lg, wlgT, 256, 512);
    transpose_bf16_kernel<<<(256 * 512 + 255) / 256, 256, 0, stream>>>(w_cg, wcgT, 256, 512);
    transpose_bf16_kernel<<<(256 * 512 + 255) / 256, 256, 0, stream>>>(w_bg, wbgT, 256, 512);
    transpose_bf16_kernel<<<(512 * 256 + 255) / 256, 256, 0, stream>>>(w_out, woutT, 512, 256);
    convert_bf16_kernel<<<(Nn * Dd + 255) / 256, 256, 0, stream>>>(local, xb, Nn * Dd);

    // -------- pair branch (fused features + LN + MLP + attn bias) --------
    pair_kernel<<<Nn, 32, 0, stream>>>(pos, nbr, resi, chain, mask,
                                       w_relpos, w_dist, w_dir, w_rot,
                                       ln_pair_g, ln_pair_b, wp1T, b_p1, wp2T, b_p2, wb,
                                       biasL, pmask);

    // -------- QKV projections --------
    gemm_kernel<0><<<g256, 256, 0, stream>>>(xb, wqT, nullptr, qf, nullptr, Nn, 256, 256);
    gemm_kernel<0><<<g256, 256, 0, stream>>>(xb, wkT, nullptr, kf, nullptr, Nn, 256, 256);
    gemm_kernel<0><<<g256, 256, 0, stream>>>(xb, wvT, nullptr, vf, nullptr, Nn, 256, 256);

    // -------- attention + output projection --------
    attn_kernel<<<Nn / 8, 256, 0, stream>>>(qf, kf, vf, biasL, pmask, nbr, ob);
    gemm_kernel<0><<<g256, 256, 0, stream>>>(ob, woT, nullptr, yf, nullptr, Nn, 256, 256);
    addln_kernel<<<Nn, 256, 0, stream>>>(local, yf, mask, ln_a_g, ln_a_b, x1, x1b, incf);

    // -------- tfeat branch --------
    tfeat_kernel<<<Nn, 32, 0, stream>>>(pos, nbr, pmask, tfr);
    gemm_kernel<1><<<g512, 256, 0, stream>>>(tfr, wt1T, nullptr, nullptr, t1b, Nn, 512, 608);
    gemm_kernel<0><<<g256, 256, 0, stream>>>(t1b, wt2T, nullptr, tfo, nullptr, Nn, 256, 512);

    // -------- gates --------
    gemm_kernel<0><<<g512, 256, 0, stream>>>(x1b, wupT, nullptr, upf, nullptr, Nn, 512, 256);
    gemm_kernel<1><<<g512, 256, 0, stream>>>(x1b, wlgT, nullptr, lgf, nullptr, Nn, 512, 256);
    gemm_kernel<1><<<g512, 256, 0, stream>>>(x1b, wcgT, nullptr, cgf, nullptr, Nn, 512, 256);
    gemm_kernel<1><<<g512, 256, 0, stream>>>(x1b, wbgT, nullptr, bgf, nullptr, Nn, 512, 256);

    zero_kernel<<<(segTotal + 255) / 256, 256, 0, stream>>>(segbuf, segTotal);
    seg_accum_kernel<<<Nn, 256, 0, stream>>>(upf, batch, chain, mask, sumB, cntB, sumC, cntC);
    seg_div_kernel<<<((NBATCH + NCHAIN) * 512 + 255) / 256, 256, 0, stream>>>(sumB, cntB, sumC, cntC);
    gate_combine_kernel<<<(Nn * 512 + 255) / 256, 256, 0, stream>>>(upf, lgf, cgf, bgf, sumB, sumC,
                                                                    batch, chain, hb);
    gemm_kernel<0><<<g256, 256, 0, stream>>>(hb, woutT, b_out, y2g, nullptr, Nn, 256, 512);

    // -------- final residual + double LayerNorm --------
    final_kernel<<<Nn, 256, 0, stream>>>(x1, y2g, tfo, mask, incf,
                                         ln_u_g, ln_u_b, ln_f_g, ln_f_b, outp);
}
